// GraphAttention_70351564308525
// MI455X (gfx1250) — compile-verified, matrix operand order checked by reading, not run on hardware
//
#include <hip/hip_runtime.h>
#include <hip/hip_bf16.h>

// GAT fused kernels for MI455X (gfx1250, wave32, WMMA 16x16x32 f16,
// async global->LDS double buffering for the adjacency stream).
// N=4096, IN_F=256, OUT_F=64, HEADS=8, F = HEADS*OUT_F = 512.

#define N_NODES 4096
#define IN_F    256
#define OUT_F   64
#define HEADS   8
#define F_TOT   512
#define NEG_SLOPE 0.2f
#define MASK_VAL  -9.0e15f
#define ADJ_PITCH 36   // 16 rows x 36 floats: 36*mr mod 64 distinct -> no bank conflicts

typedef __attribute__((ext_vector_type(16))) _Float16 v16h;
typedef __attribute__((ext_vector_type(8)))  float    v8f;

// WMMA 16-bit A/B fragment K index (CDNA5 ISA 7.12.2): element e of lane
// (grp = lane>>4) holds K = (e<8 ? e : e+8) + 8*grp.
__device__ __forceinline__ int frag_k(int e, int grp) {
    return ((e < 8) ? e : e + 8) + (grp << 3);
}

// gfx1250 async global->LDS copy, 16B per lane, tracked by ASYNCcnt.
__device__ __forceinline__ void async_copy_b128(unsigned lds_off, const float* gaddr) {
    asm volatile("global_load_async_to_lds_b128 %0, %1, off"
                 :: "v"(lds_off), "v"(gaddr) : "memory");
}
__device__ __forceinline__ void wait_async0() {
    asm volatile("s_wait_asynccnt 0x0" ::: "memory");
}

// =====================================================================
// Kernel 1: h = X @ W  (4096x256 @ 256x512), WMMA f16, f32 accumulate.
// One wave per 16x16 tile; emits f32 h (for coeffs) and f16 h (PV WMMA).
// =====================================================================
__global__ __launch_bounds__(128) void gat_gemm_h(
    const float* __restrict__ X, const float* __restrict__ W,
    float* __restrict__ hf32, _Float16* __restrict__ hf16)
{
    const int lane = threadIdx.x & 31;
    const int wave = threadIdx.x >> 5;
    const int tile = blockIdx.x * 4 + wave;       // 0..8191
    const int mt = tile >> 5, nt = tile & 31;
    const int grp = lane >> 4, mr = lane & 15;

    v8f c = {};
    for (int kt = 0; kt < IN_F; kt += 32) {
        v16h a, b;
#pragma unroll
        for (int e = 0; e < 16; ++e) {
            const int k = kt + frag_k(e, grp);
            a[e] = (_Float16)X[(mt * 16 + mr) * IN_F + k];
            b[e] = (_Float16)W[k * F_TOT + nt * 16 + mr];
        }
        c = __builtin_amdgcn_wmma_f32_16x16x32_f16(
                false, a, false, b, (short)0, c, false, false);
    }
#pragma unroll
    for (int e = 0; e < 8; ++e) {
        const int row = mt * 16 + e + (grp << 3);
        const int col = nt * 16 + mr;
        hf32[row * F_TOT + col] = c[e];
        hf16[row * F_TOT + col] = (_Float16)c[e];
    }
}

// =====================================================================
// Kernel 2: rank-1 logit coefficients (h is L2-resident, 8 MB << 192 MB).
// =====================================================================
__global__ __launch_bounds__(256) void gat_coeff(
    const float* __restrict__ hf32,
    const float* __restrict__ a_i, const float* __restrict__ a_j,
    float* __restrict__ ci, float* __restrict__ cj)
{
    const int idx = blockIdx.x * blockDim.x + threadIdx.x;   // n*8 + head
    if (idx >= N_NODES * HEADS) return;
    const int row = idx >> 3, head = idx & 7;
    const float* hp = hf32 + row * F_TOT + head * OUT_F;
    float si = 0.f, sj = 0.f;
#pragma unroll 8
    for (int d = 0; d < OUT_F; ++d) {
        const float v = hp[d];
        si += v * a_i[head * OUT_F + d];
        sj += v * a_j[head * OUT_F + d];
    }
    ci[idx] = si;
    cj[idx] = sj;
}

// =====================================================================
// Kernel 3: fused masked-softmax attention + aggregation (flash style).
// grid = 64 i-groups x 8 heads.  Block = 4 waves; wave w owns i-tile
// (group*4 + w) and ALL 4 d-tiles of the head (4 WMMAs share one A
// fragment -> reuse_a hints).  cj[:,head] staged once in LDS (16 KB);
// each wave double-buffers its 16x32 adj tile in LDS via
// global_load_async_to_lds_b128 issued one step ahead (ASYNCcnt).
// e[N,N,H] is never materialized: logits are rank-1 (ci[i]+cj[j]).
// =====================================================================
__global__ __launch_bounds__(128) void gat_attn(
    const float* __restrict__ adj, const _Float16* __restrict__ hf16,
    const float* __restrict__ ci, const float* __restrict__ cj,
    const float* __restrict__ bias, float* __restrict__ out)
{
    __shared__ float cj_sh[N_NODES];                      // 16 KB
    __shared__ float adj_sh[4][2][16 * ADJ_PITCH];        // 18 KB (4 waves x 2 bufs)

    const int lane  = threadIdx.x & 31;
    const int wave  = threadIdx.x >> 5;
    const int head  = blockIdx.x & 7;
    const int itile = (blockIdx.x >> 3) * 4 + wave;       // 0..255
    const int grp   = lane >> 4;
    const int mr    = lane & 15;
    const int i0    = itile * 16;

    // stage this head's cj column into LDS (stride-8 gather, once per block)
    for (int j = threadIdx.x; j < N_NODES; j += 128)
        cj_sh[j] = cj[j * HEADS + head];
    __syncthreads();

    const float  ci_l    = ci[(i0 + mr) * HEADS + head];      // row term, row mr
    const float* adj_row = adj + (size_t)(i0 + mr) * N_NODES; // lane's adj row
    float* const adj_dst = &adj_sh[wave][0][0];

    // async stage of one 16x32 fp32 tile: lane (grp,mr) copies 16 floats of
    // row mr, columns j0 + grp*16 .. +15, as 4x b128.
    auto stage = [&](int buf, int j0) {
#pragma unroll
        for (int q = 0; q < 4; ++q) {
            float* lp = adj_dst + buf * (16 * ADJ_PITCH) + mr * ADJ_PITCH
                        + grp * 16 + q * 4;
            async_copy_b128((unsigned)(uintptr_t)lp, adj_row + j0 + grp * 16 + q * 4);
        }
    };

    stage(0, 0);

    float mrow = -3.0e38f, lrow = 0.0f;       // online softmax state for row mr
    v8f acc0 = {}, acc1 = {}, acc2 = {}, acc3 = {};
    const _Float16* hb = hf16 + head * OUT_F + mr;   // + j*F_TOT + dt*16

    for (int j0 = 0; j0 < N_NODES; j0 += 32) {
        const int cur = (j0 >> 5) & 1;
        wait_async0();                               // tile 'cur' landed in LDS
        if (j0 + 32 < N_NODES) stage(cur ^ 1, j0 + 32);
        const float* at = adj_dst + cur * (16 * ADJ_PITCH) + mr * ADJ_PITCH;

        // ---- logits for 16x32 tile in WMMA-A order ----
        float s[16];
        float lmax = -3.0e38f;
#pragma unroll
        for (int e = 0; e < 16; ++e) {
            const int k = frag_k(e, grp);
            float v = ci_l + cj_sh[j0 + k];
            v = (v >= 0.f) ? v : NEG_SLOPE * v;      // leaky relu
            v = (at[k] > 0.f) ? v : MASK_VAL;        // adjacency mask (from LDS)
            s[e] = v;
            lmax = fmaxf(lmax, v);
        }
        lmax = fmaxf(lmax, __shfl_xor(lmax, 16, 32));      // lane pair -> row max
        const float mnew  = fmaxf(mrow, lmax);
        const float scale = __expf(mrow - mnew);

        // ---- probabilities -> f16 A fragment ----
        v16h a;
        float psum = 0.f;
#pragma unroll
        for (int e = 0; e < 16; ++e) {
            const float p = __expf(s[e] - mnew);
            a[e] = (_Float16)p;
            psum += p;
        }
        psum += __shfl_xor(psum, 16, 32);
        lrow = lrow * scale + psum;
        mrow = mnew;

        // ---- B fragments: h rows j0..j0+31, 4 d-tiles of this head ----
        v16h b0, b1, b2, b3;
#pragma unroll
        for (int e = 0; e < 16; ++e) {
            const _Float16* hr = hb + (size_t)(j0 + frag_k(e, grp)) * F_TOT;
            b0[e] = hr[0];  b1[e] = hr[16];  b2[e] = hr[32];  b3[e] = hr[48];
        }

        // ---- per-row accumulator rescale (C row = e + 8*grp) ----
#pragma unroll
        for (int e = 0; e < 8; ++e) {
            const float f = __shfl(scale, e + (grp << 3), 32);
            acc0[e] *= f; acc1[e] *= f; acc2[e] *= f; acc3[e] *= f;
        }

        // 4 WMMAs share A: set reuse-A hint on the repeats (ISA 7.12, OPSEL.RA)
        acc0 = __builtin_amdgcn_wmma_f32_16x16x32_f16(false, a, false, b0, (short)0, acc0, false, false);
        acc1 = __builtin_amdgcn_wmma_f32_16x16x32_f16(false, a, false, b1, (short)0, acc1, true,  false);
        acc2 = __builtin_amdgcn_wmma_f32_16x16x32_f16(false, a, false, b2, (short)0, acc2, true,  false);
        acc3 = __builtin_amdgcn_wmma_f32_16x16x32_f16(false, a, false, b3, (short)0, acc3, true,  false);
    }

    // ---- finalize: /rowsum, +bias, store ----
    const int cbase = head * OUT_F + mr;
#pragma unroll
    for (int e = 0; e < 8; ++e) {
        const int r = e + (grp << 3);
        const float linv = 1.0f / __shfl(lrow, r, 32);
        float* orow = out + (size_t)(i0 + r) * F_TOT + cbase;
        orow[0]  = acc0[e] * linv + bias[cbase];
        orow[16] = acc1[e] * linv + bias[cbase + 16];
        orow[32] = acc2[e] * linv + bias[cbase + 32];
        orow[48] = acc3[e] * linv + bias[cbase + 48];
    }
}

// =====================================================================
// Host-side launch. Workspace (~12.3 MB):
//   [0,8MB) h fp32 | [8MB,12MB) h fp16 | [12MB,..) ci, cj (128 KB each)
// =====================================================================
extern "C" void kernel_launch(void* const* d_in, const int* in_sizes, int n_in,
                              void* d_out, int out_size, void* d_ws, size_t ws_size,
                              hipStream_t stream) {
    const float* X    = (const float*)d_in[0];
    const float* adj  = (const float*)d_in[1];
    const float* W    = (const float*)d_in[2];
    const float* a_i  = (const float*)d_in[3];
    const float* a_j  = (const float*)d_in[4];
    const float* bias = (const float*)d_in[5];
    float* out = (float*)d_out;

    char* ws = (char*)d_ws;
    float*    hf32 = (float*)ws;
    _Float16* hf16 = (_Float16*)(ws + (size_t)8  * (1 << 20));
    float*    ci   = (float*)  (ws + (size_t)12 * (1 << 20));
    float*    cj   = ci + N_NODES * HEADS;

    gat_gemm_h<<<2048, 128, 0, stream>>>(X, W, hf32, hf16);
    gat_coeff<<<(N_NODES * HEADS + 255) / 256, 256, 0, stream>>>(hf32, a_i, a_j, ci, cj);
    gat_attn<<<64 * HEADS, 128, 0, stream>>>(adj, hf16, ci, cj, bias, out);
}
